// SwinMoBABlock_14276471292735
// MI455X (gfx1250) — matrix operand order repaired and checked by reading, not compiled, not run on hardware
//
#include <hip/hip_runtime.h>
#include <hip/hip_bf16.h>
#include <math.h>

// ---------------------------------------------------------------------------
// SwinMoBABlock for MI455X (gfx1250): WMMA GEMMs + async global->LDS staging.
// ---------------------------------------------------------------------------

typedef _Float16 h16;
typedef __attribute__((ext_vector_type(16))) _Float16 v16h;
typedef __attribute__((ext_vector_type(8)))  _Float16 v8h;
typedef __attribute__((ext_vector_type(8)))  float    v8f;
typedef int i4v __attribute__((__vector_size__(4 * sizeof(int))));

static constexpr int   TOK   = 25088;                 // 8 * 3136 tokens
static constexpr float SCALE = 0.17677669529663687f;  // 32^-0.5

// ---- workspace layout (bytes) ---------------------------------------------
static constexpr size_t SZ_XN  = (size_t)TOK * 256 * 2;
static constexpr size_t SZ_Q   = (size_t)TOK * 256 * 2;
static constexpr size_t SZ_K   = (size_t)32 * 8 * 49 * 32 * 2;
static constexpr size_t SZ_V   = SZ_K;
static constexpr size_t SZ_O   = (size_t)TOK * 256 * 2;
static constexpr size_t SZ_H1  = (size_t)TOK * 1024 * 2;
static constexpr size_t SZ_X2  = (size_t)TOK * 256 * 4;

static constexpr size_t OFF_XN = 0;
static constexpr size_t OFF_Q  = OFF_XN + SZ_XN;
static constexpr size_t OFF_K  = OFF_Q + SZ_Q;
static constexpr size_t OFF_V  = OFF_K + SZ_K;
static constexpr size_t OFF_O  = OFF_V + SZ_V;
static constexpr size_t OFF_M  = 0;                  // reuses XN (dead)
static constexpr size_t OFF_H1 = SZ_XN;              // reuses Q/K/V/O (dead)
static constexpr size_t OFF_X2 = OFF_H1 + SZ_H1;
static constexpr size_t OFF_WQKV = OFF_X2 + SZ_X2;   // persistent f16 weights
static constexpr size_t OFF_WPRJ = OFF_WQKV + (size_t)768 * 256 * 2;
static constexpr size_t OFF_WFC1 = OFF_WPRJ + (size_t)256 * 256 * 2;
static constexpr size_t OFF_WFC2 = OFF_WFC1 + (size_t)1024 * 256 * 2;

// ---- async global->LDS (CDNA5) --------------------------------------------
#if defined(__gfx1250__) && __has_builtin(__builtin_amdgcn_global_load_async_to_lds_b128)
#define USE_ASYNC_LDS 1
#else
#define USE_ASYNC_LDS 0
#endif

#if USE_ASYNC_LDS
typedef __attribute__((address_space(1))) i4v glob_i4v;
typedef __attribute__((address_space(3))) i4v lds_i4v;
#endif

// copy 16 bytes global -> LDS
__device__ __forceinline__ void stage16(const h16* g, h16* l) {
#if USE_ASYNC_LDS
  __builtin_amdgcn_global_load_async_to_lds_b128(
      (glob_i4v*)g, (lds_i4v*)l, 0, 0);
#else
  *(v8h*)l = *(const v8h*)g;
#endif
}

__device__ __forceinline__ void wait_async_all() {
#if USE_ASYNC_LDS
#if __has_builtin(__builtin_amdgcn_s_wait_asynccnt)
  __builtin_amdgcn_s_wait_asynccnt(0);
#else
  asm volatile("s_wait_asynccnt 0x0" ::: "memory");
#endif
#endif
}

// ---- WMMA helpers ----------------------------------------------------------
__device__ __forceinline__ v8f wmma_f16(v16h a, v16h b, v8f c) {
  return __builtin_amdgcn_wmma_f32_16x16x32_f16(false, a, false, b,
                                                (short)0, c, false, false);
}

// A operand 16x32 f16 from LDS row-major tile (ld halves).
// lane<16: K{0..7,16..23}, lane>=16: K{8..15,24..31}, row = lane&15.
__device__ __forceinline__ v16h load_a_frag(const h16* base, int ld) {
  const int lane = threadIdx.x & 31;
  const h16* p = base + (lane & 15) * ld + ((lane >> 4) << 3);
  v8h lo = *(const v8h*)(p);
  v8h hi = *(const v8h*)(p + 16);
  v16h r;
#pragma unroll
  for (int i = 0; i < 8; ++i) { r[i] = lo[i]; r[i + 8] = hi[i]; }
  return r;
}

// B operand 32x16 f16 from LDS tile stored B^T[n][k] (ld halves per n-row).
// lane<16: K{0..7,16..23}, lane>=16: K{8..15,24..31}, column n = lane&15.
__device__ __forceinline__ v16h load_b_frag(const h16* base, int ld) {
  const int lane = threadIdx.x & 31;
  const h16* p = base + (lane & 15) * ld + ((lane >> 4) << 3);
  v8h lo = *(const v8h*)(p);
  v8h hi = *(const v8h*)(p + 16);
  v16h r;
#pragma unroll
  for (int i = 0; i < 8; ++i) { r[i] = lo[i]; r[i + 8] = hi[i]; }
  return r;
}

// ---- weight convert + transpose: f32 (K x N) -> f16 (N x K) ----------------
__global__ __launch_bounds__(256) void convert_transpose_kernel(
    const float* __restrict__ in, h16* __restrict__ out, int K, int N) {
  int idx = blockIdx.x * 256 + threadIdx.x;
  if (idx >= K * N) return;
  int k = idx / N, n = idx - k * N;
  out[(size_t)n * K + k] = (h16)in[idx];
}

// ---- LayerNorm (one 256-wide token row per block), optional window permute -
__global__ __launch_bounds__(256) void ln_kernel(
    const float* __restrict__ x, const float* __restrict__ g,
    const float* __restrict__ bta, h16* __restrict__ out, int permute) {
  __shared__ float red[256];
  const int row = blockIdx.x;
  const int c = threadIdx.x;
  float v = x[(size_t)row * 256 + c];
  red[c] = v;
  __syncthreads();
  for (int s = 128; s > 0; s >>= 1) { if (c < s) red[c] += red[c + s]; __syncthreads(); }
  float mu = red[0] * (1.f / 256.f);
  __syncthreads();
  float d = v - mu;
  red[c] = d * d;
  __syncthreads();
  for (int s = 128; s > 0; s >>= 1) { if (c < s) red[c] += red[c + s]; __syncthreads(); }
  float var = red[0] * (1.f / 256.f);
  float y = d * rsqrtf(var + 1e-5f) * g[c] + bta[c];
  size_t orow = row;
  if (permute) {                        // token (b,l) -> window row (b*64+w)*49+n
    int b = row / 3136, l = row - b * 3136;
    int r = l / 56, cc = l - r * 56;
    int wInd = (r / 7) * 8 + (cc / 7);
    int n = (r % 7) * 7 + (cc % 7);
    orow = ((size_t)b * 64 + wInd) * 49 + n;
  }
  out[orow * 256 + c] = (h16)y;
}

// ---- double-buffered 64x64 WMMA GEMM core (K multiple of 32, 128 thr) ------
static constexpr int TBUF = 64 * 32;   // halves per LDS tile buffer

__device__ __forceinline__ void stage_tiles(const h16* asrc, h16* adst,
                                            const h16* bsrc, h16* bdst) {
#if USE_ASYNC_LDS
  stage16(asrc, adst);
  stage16(asrc + 8, adst + 8);
  stage16(bsrc, bdst);
  stage16(bsrc + 8, bdst + 8);
#else
  v8h a0 = *(const v8h*)(asrc);
  v8h a1 = *(const v8h*)(asrc + 8);
  v8h b0 = *(const v8h*)(bsrc);
  v8h b1 = *(const v8h*)(bsrc + 8);
  *(v8h*)(adst) = a0; *(v8h*)(adst + 8) = a1;
  *(v8h*)(bdst) = b0; *(v8h*)(bdst + 8) = b1;
#endif
}

__device__ __forceinline__ void gemm_core_64x64(
    const h16* __restrict__ A, int lda, const h16* __restrict__ Wt, int K,
    int m0, int n0, h16* As, h16* Bs, v8f acc[4]) {
  const int tid = threadIdx.x;
  const int wv  = tid >> 5;
  const int row = tid >> 1, seg = tid & 1;
  const h16* asrc = A + (size_t)(m0 + row) * lda + seg * 16;
  const h16* bsrc = Wt + (size_t)(n0 + row) * K + seg * 16;
  h16* adst = As + row * 32 + seg * 16;
  h16* bdst = Bs + row * 32 + seg * 16;

  stage_tiles(asrc, adst, bsrc, bdst);          // prologue: tile 0 -> buf 0
  int cur = 0;
  for (int k0 = 0; k0 < K; k0 += 32) {
    wait_async_all();
    __syncthreads();                            // tile `cur` resident
    const int nxt = cur ^ 1;
    if (k0 + 32 < K)                            // prefetch next tile
      stage_tiles(asrc + k0 + 32, adst + nxt * TBUF,
                  bsrc + k0 + 32, bdst + nxt * TBUF);
    v16h a = load_a_frag(As + cur * TBUF + wv * 16 * 32, 32);
#pragma unroll
    for (int j = 0; j < 4; ++j) {
      v16h b = load_b_frag(Bs + cur * TBUF + j * 16 * 32, 32);
      acc[j] = wmma_f16(a, b, acc[j]);
    }
    cur = nxt;
  }
}

// ---- QKV GEMM: xn (window rows, 25088x256) x Wt(768x256) + bias ------------
__global__ __launch_bounds__(128) void qkv_gemm_kernel(
    const h16* __restrict__ A, const h16* __restrict__ Wt,
    const float* __restrict__ bias, h16* __restrict__ qb,
    h16* __restrict__ kb, h16* __restrict__ vb) {
  __shared__ __align__(16) h16 As[2 * TBUF];
  __shared__ __align__(16) h16 Bs[2 * TBUF];
  v8f acc[4] = {};
  const int m0 = blockIdx.x * 64, n0 = blockIdx.y * 64;
  gemm_core_64x64(A, 256, Wt, 256, m0, n0, As, Bs, acc);
  const int wv = threadIdx.x >> 5, lane = threadIdx.x & 31;
  const int mBase = m0 + wv * 16 + ((lane >> 4) << 3);
  const int nCol = lane & 15;
#pragma unroll
  for (int j = 0; j < 4; ++j) {
    int c = n0 + j * 16 + nCol;
    float bv = bias[c];
    int part = c >> 8, hc = (c >> 5) & 7, d = c & 31;
#pragma unroll
    for (int r = 0; r < 8; ++r) {
      int row = mBase + r;
      int g = row / 49, n = row - g * 49;
      h16 hv = (h16)(acc[j][r] + bv);
      if (part == 0) {
        qb[(((size_t)g * 8 + hc) * 49 + n) * 32 + d] = hv;
      } else if (g < 32) {                 // only batch-0 windows 0..31 feed K/V
        if (part == 1) kb[(((size_t)g * 8 + hc) * 49 + n) * 32 + d] = hv;
        else           vb[(((size_t)g * 8 + hc) * 32 + d) * 49 + n] = hv;  // V^T
      }
    }
  }
}

// ---- attention: out[f'] = 1/4 * sum_t Attn(q[t*128+f'/4], kv[f' mod 32]) ---
// Q/K staged as flat 16B-chunk copies (49*32 halves = 196 chunks); pad rows
// 49..63 hold garbage which provably never reaches a stored output element.
__global__ __launch_bounds__(128) void attn_kernel(
    const h16* __restrict__ q, const h16* __restrict__ k,
    const h16* __restrict__ v, const float* __restrict__ rpb,
    h16* __restrict__ o) {
  const int fp = blockIdx.x >> 3;   // output window f' in [0,512)
  const int h  = blockIdx.x & 7;
  const int tid = threadIdx.x;
  const int wv = tid >> 5, lane = tid & 31;

  __shared__ __align__(16) h16  Qs[64 * 32];
  __shared__ __align__(16) h16  Ks[64 * 32];
  __shared__ __align__(16) h16  Vs[32 * 64];     // V^T[d][key], pad keys zeroed
  __shared__ __align__(16) float Ss[64 * 64];
  __shared__ __align__(16) h16  Ps[64 * 64];

  const int mwin = fp & 31;                      // the single KV window
  { // stage K once: flat copy of 196 x 16B
    const h16* src = k + ((size_t)mwin * 8 + h) * 49 * 32;
    stage16(src + tid * 8, Ks + tid * 8);
    if (tid < 68) stage16(src + (128 + tid) * 8, Ks + (128 + tid) * 8);
  }
  if (tid < 64) { // stage V^T (keys 49..63 explicitly zero)
    int d = tid >> 1, seg = tid & 1;
    const h16* src = v + (((size_t)mwin * 8 + h) * 32 + d) * 49;
    for (int j = 0; j < 32; ++j) {
      int key = seg * 32 + j;
      Vs[d * 64 + key] = (key < 49) ? src[key] : (h16)0.f;
    }
  }

  v8f accO[2] = {};
  const int fq4 = fp >> 2;
  for (int t = 0; t < 4; ++t) {
    const int g = t * 128 + fq4;                 // query window for this term
    { // stage Q for this term: flat copy of 196 x 16B
      const h16* src = q + ((size_t)g * 8 + h) * 49 * 32;
      stage16(src + tid * 8, Qs + tid * 8);
      if (tid < 68) stage16(src + (128 + tid) * 8, Qs + (128 + tid) * 8);
    }
    wait_async_all();
    __syncthreads();                             // Qs (and on t=0: Ks/Vs) ready
    // S = Q K^T  (64x64 padded, K-dim = 32: exactly one WMMA step)
    v16h a = load_a_frag(Qs + wv * 16 * 32, 32);
#pragma unroll
    for (int nt = 0; nt < 4; ++nt) {
      v16h bf = load_b_frag(Ks + nt * 16 * 32, 32);
      v8f s = {};
      s = wmma_f16(a, bf, s);
      int mr = wv * 16 + ((lane >> 4) << 3);
      int nc = nt * 16 + (lane & 15);
#pragma unroll
      for (int r = 0; r < 8; ++r) Ss[(mr + r) * 64 + nc] = s[r];
    }
    __syncthreads();
    // softmax with scale + on-the-fly relative position bias
    if (tid < 64) {
      int row = tid;
      int rh = row / 7, rw = row % 7;
      float mx = -1e30f;
      for (int n = 0; n < 49; ++n) {
        float bias = 0.f;
        if (row < 49) {
          int nh = n / 7, nw = n % 7;
          bias = rpb[((rh - nh + 6) * 13 + (rw - nw + 6)) * 8 + h];
        }
        float l = Ss[row * 64 + n] * SCALE + bias;
        Ss[row * 64 + n] = l;
        mx = fmaxf(mx, l);
      }
      float sum = 0.f;
      for (int n = 0; n < 49; ++n) {
        float e = __expf(Ss[row * 64 + n] - mx);
        Ss[row * 64 + n] = e;
        sum += e;
      }
      float inv = 1.f / sum;
      for (int n = 0; n < 49; ++n) Ps[row * 64 + n] = (h16)(Ss[row * 64 + n] * inv);
      for (int n = 49; n < 64; ++n) Ps[row * 64 + n] = (h16)0.f;
    }
    __syncthreads();
    // O += P V   (K-dim 64 padded -> 2 steps; N = 32 dims -> 2 tiles)
#pragma unroll
    for (int ks = 0; ks < 2; ++ks) {
      v16h ap = load_a_frag(Ps + wv * 16 * 64 + ks * 32, 64);
#pragma unroll
      for (int nt2 = 0; nt2 < 2; ++nt2) {
        v16h bv = load_b_frag(Vs + nt2 * 16 * 64 + ks * 32, 64);
        accO[nt2] = wmma_f16(ap, bv, accO[nt2]);
      }
    }
  }
  // store mean (1/4), window-row layout (f', n, C) with head slice
  const int mr = wv * 16 + ((lane >> 4) << 3);
  const int nc = lane & 15;
#pragma unroll
  for (int nt2 = 0; nt2 < 2; ++nt2)
#pragma unroll
    for (int r = 0; r < 8; ++r) {
      int mm = mr + r;
      if (mm < 49)
        o[((size_t)fp * 49 + mm) * 256 + h * 32 + nt2 * 16 + nc] =
            (h16)(accO[nt2][r] * 0.25f);
    }
}

// ---- proj GEMM + window-reverse + residual -> x2 (f32, token order) --------
__global__ __launch_bounds__(128) void proj_gemm_kernel(
    const h16* __restrict__ A, const h16* __restrict__ Wt,
    const float* __restrict__ bias, const float* __restrict__ xin,
    float* __restrict__ x2) {
  __shared__ __align__(16) h16 As[2 * TBUF];
  __shared__ __align__(16) h16 Bs[2 * TBUF];
  v8f acc[4] = {};
  const int m0 = blockIdx.x * 64, n0 = blockIdx.y * 64;
  gemm_core_64x64(A, 256, Wt, 256, m0, n0, As, Bs, acc);
  const int wv = threadIdx.x >> 5, lane = threadIdx.x & 31;
  const int mBase = m0 + wv * 16 + ((lane >> 4) << 3);
  const int nCol = lane & 15;
#pragma unroll
  for (int j = 0; j < 4; ++j) {
    int c = n0 + j * 16 + nCol;
    float bv = bias[c];
#pragma unroll
    for (int r = 0; r < 8; ++r) {
      int row = mBase + r;
      int g = row / 49, n = row - g * 49;
      int b2 = g >> 6, wInd = g & 63;
      int ri = (wInd >> 3) * 7 + n / 7;
      int ci = (wInd & 7) * 7 + n % 7;
      size_t trow = (size_t)b2 * 3136 + ri * 56 + ci;
      x2[trow * 256 + c] = acc[j][r] + bv + xin[trow * 256 + c];
    }
  }
}

// ---- FC1 GEMM + exact GELU -> h1 f16 ---------------------------------------
__global__ __launch_bounds__(128) void fc1_gemm_kernel(
    const h16* __restrict__ A, const h16* __restrict__ Wt,
    const float* __restrict__ bias, h16* __restrict__ h1) {
  __shared__ __align__(16) h16 As[2 * TBUF];
  __shared__ __align__(16) h16 Bs[2 * TBUF];
  v8f acc[4] = {};
  const int m0 = blockIdx.x * 64, n0 = blockIdx.y * 64;
  gemm_core_64x64(A, 256, Wt, 256, m0, n0, As, Bs, acc);
  const int wv = threadIdx.x >> 5, lane = threadIdx.x & 31;
  const int mBase = m0 + wv * 16 + ((lane >> 4) << 3);
  const int nCol = lane & 15;
#pragma unroll
  for (int j = 0; j < 4; ++j) {
    int c = n0 + j * 16 + nCol;
    float bv = bias[c];
#pragma unroll
    for (int r = 0; r < 8; ++r) {
      float gv = acc[j][r] + bv;
      float gel = 0.5f * gv * (1.f + erff(gv * 0.7071067811865476f));
      h1[(size_t)(mBase + r) * 1024 + c] = (h16)gel;
    }
  }
}

// ---- FC2 GEMM + bias + residual(x2) -> out f32 -----------------------------
__global__ __launch_bounds__(128) void fc2_gemm_kernel(
    const h16* __restrict__ A, const h16* __restrict__ Wt,
    const float* __restrict__ bias, const float* __restrict__ x2,
    float* __restrict__ out) {
  __shared__ __align__(16) h16 As[2 * TBUF];
  __shared__ __align__(16) h16 Bs[2 * TBUF];
  v8f acc[4] = {};
  const int m0 = blockIdx.x * 64, n0 = blockIdx.y * 64;
  gemm_core_64x64(A, 1024, Wt, 1024, m0, n0, As, Bs, acc);
  const int wv = threadIdx.x >> 5, lane = threadIdx.x & 31;
  const int mBase = m0 + wv * 16 + ((lane >> 4) << 3);
  const int nCol = lane & 15;
#pragma unroll
  for (int j = 0; j < 4; ++j) {
    int c = n0 + j * 16 + nCol;
    float bv = bias[c];
#pragma unroll
    for (int r = 0; r < 8; ++r) {
      size_t rr = (size_t)(mBase + r);
      out[rr * 256 + c] = acc[j][r] + bv + x2[rr * 256 + c];
    }
  }
}

// ---------------------------------------------------------------------------
extern "C" void kernel_launch(void* const* d_in, const int* in_sizes, int n_in,
                              void* d_out, int out_size, void* d_ws, size_t ws_size,
                              hipStream_t stream) {
  (void)in_sizes; (void)n_in; (void)out_size; (void)ws_size;
  const float* x      = (const float*)d_in[0];
  const float* n1g    = (const float*)d_in[1];
  const float* n1b    = (const float*)d_in[2];
  const float* qkv_w  = (const float*)d_in[3];
  const float* qkv_b  = (const float*)d_in[4];
  const float* rpb    = (const float*)d_in[5];
  const float* proj_w = (const float*)d_in[6];
  const float* proj_b = (const float*)d_in[7];
  const float* n2g    = (const float*)d_in[8];
  const float* n2b    = (const float*)d_in[9];
  const float* fc1_w  = (const float*)d_in[10];
  const float* fc1_b  = (const float*)d_in[11];
  const float* fc2_w  = (const float*)d_in[12];
  const float* fc2_b  = (const float*)d_in[13];
  float* out = (float*)d_out;
  char* ws = (char*)d_ws;

  h16* xn   = (h16*)(ws + OFF_XN);
  h16* qb   = (h16*)(ws + OFF_Q);
  h16* kb   = (h16*)(ws + OFF_K);
  h16* vb   = (h16*)(ws + OFF_V);
  h16* ob   = (h16*)(ws + OFF_O);
  h16* mb   = (h16*)(ws + OFF_M);
  h16* h1   = (h16*)(ws + OFF_H1);
  float* x2 = (float*)(ws + OFF_X2);
  h16* wqkv = (h16*)(ws + OFF_WQKV);
  h16* wprj = (h16*)(ws + OFF_WPRJ);
  h16* wfc1 = (h16*)(ws + OFF_WFC1);
  h16* wfc2 = (h16*)(ws + OFF_WFC2);

  // weights f32 (K x N) -> f16 transposed (N x K)
  convert_transpose_kernel<<<(256 * 768 + 255) / 256, 256, 0, stream>>>(qkv_w, wqkv, 256, 768);
  convert_transpose_kernel<<<(256 * 256 + 255) / 256, 256, 0, stream>>>(proj_w, wprj, 256, 256);
  convert_transpose_kernel<<<(256 * 1024 + 255) / 256, 256, 0, stream>>>(fc1_w, wfc1, 256, 1024);
  convert_transpose_kernel<<<(1024 * 256 + 255) / 256, 256, 0, stream>>>(fc2_w, wfc2, 1024, 256);

  // LN1 + window partition
  ln_kernel<<<TOK, 256, 0, stream>>>(x, n1g, n1b, xn, 1);
  // QKV projection
  qkv_gemm_kernel<<<dim3(TOK / 64, 12), 128, 0, stream>>>(xn, wqkv, qkv_b, qb, kb, vb);
  // window attention with faithful dead-MoBA routing + RPB softmax
  attn_kernel<<<512 * 8, 128, 0, stream>>>(qb, kb, vb, rpb, ob);
  // proj + window reverse + residual
  proj_gemm_kernel<<<dim3(TOK / 64, 4), 128, 0, stream>>>(ob, wprj, proj_b, x, x2);
  // LN2
  ln_kernel<<<TOK, 256, 0, stream>>>(x2, n2g, n2b, mb, 0);
  // MLP
  fc1_gemm_kernel<<<dim3(TOK / 64, 16), 128, 0, stream>>>(mb, wfc1, fc1_b, h1);
  fc2_gemm_kernel<<<dim3(TOK / 64, 4), 128, 0, stream>>>(h1, wfc2, fc2_b, x2, out);
}
